// SparseConv3d_20229295964829
// MI455X (gfx1250) — matrix operand order repaired and checked by reading, not compiled, not run on hardware
//
#include <hip/hip_runtime.h>

// Submanifold sparse 3x3x3 conv, output-stationary WMMA formulation.
// out[i,:] = bias + sum_k mask[k,i] * feats[in_idx[k,i],:] @ weight[k]
// (out_idx is the identity wherever mask==1 -> no scatter/atomics.)
//
// Fast path (ws >= ~33.8MB):
//   pre-kernel 1: weights f32 -> f16, pre-swizzled to CDNA5 B-matrix layout
//   pre-kernel 2: feats  f32 -> f16 table (+ zero row at index ntot)
//   main kernel : no LDS, no barriers, no per-iteration cvt/mul VALU;
//                 mask applied by redirecting the gather to the zero row;
//                 wave-uniform ballot skips offsets with an all-empty tile;
//                 next row prefetched with global_prefetch_b8 (no LOADcnt).

typedef __attribute__((ext_vector_type(16))) _Float16 v16h;
typedef __attribute__((ext_vector_type(8)))  _Float16 v8h;
typedef __attribute__((ext_vector_type(8)))  float    v8f;
typedef __attribute__((ext_vector_type(4)))  float    v4f;

#define C_IN   64
#define C_OUT  64
#define K_OFF  27
#define ROWS_PER_BLOCK 128       // 8 waves * 16 rows
#define WS_FRAGS (K_OFF * 256)   // 27 offsets * 256 fragments of 16 f16
#define WSB_BYTES ((size_t)WS_FRAGS * 32)          // 221184, 256-aligned

// ---------------------------------------------------------------------------
// Pre-kernel 1: weight[k] (f32 [64][64]) -> f16 B-swizzled fragments.
// Fragment t of offset k (t = kt*128 + h*64 + n) holds w[kt*32+8h+(j<8?j:j+8)][n], j=0..15.
// ---------------------------------------------------------------------------
__global__ __launch_bounds__(256) void convert_weights_bswizzle(
    const float* __restrict__ weight, v16h* __restrict__ wsB)
{
    const int k  = blockIdx.x;
    const int t  = threadIdx.x;
    const int kt = (t >> 7) & 1;
    const int hh = (t >> 6) & 1;
    const int n  =  t & 63;

    const float* wk = weight + (size_t)k * (C_IN * C_OUT)
                    + (size_t)(kt * 32 + hh * 8) * C_OUT + n;
    v16h wv;
    #pragma unroll
    for (int j = 0; j < 16; ++j)
        wv[j] = (_Float16)wk[(size_t)(j < 8 ? j : j + 8) * C_OUT];
    wsB[(size_t)k * 256 + t] = wv;
}

// ---------------------------------------------------------------------------
// Pre-kernel 2: feats f32 -> f16 (row-major, same layout), zero row at ntot.
// ---------------------------------------------------------------------------
__global__ __launch_bounds__(256) void convert_feats_f16(
    const float* __restrict__ feats, _Float16* __restrict__ f16, int ntot)
{
    const size_t gid  = (size_t)blockIdx.x * 256 + threadIdx.x;
    const size_t base = gid * 8;
    if (base < (size_t)ntot * C_IN) {
        const v4f a = *(const v4f*)(feats + base);
        const v4f b = *(const v4f*)(feats + base + 4);
        v8h o;
        #pragma unroll
        for (int j = 0; j < 4; ++j) { o[j] = (_Float16)a[j]; o[j+4] = (_Float16)b[j]; }
        *(v8h*)(f16 + base) = o;
    }
    if (blockIdx.x == 0 && threadIdx.x < 8) {          // zero row at index ntot
        v8h z = {};
        *(v8h*)(f16 + (size_t)ntot * C_IN + threadIdx.x * 8) = z;
    }
}

// ---------------------------------------------------------------------------
// Main kernel (fast path): pure gather + WMMA loop.
// ---------------------------------------------------------------------------
__global__ __launch_bounds__(256) void sparse_conv3d_wmma_f16tab(
    const _Float16* __restrict__ feats16,  // [N+1, 64], row N = zeros
    const v16h*     __restrict__ wsB,      // [27 * 256] B fragments
    const float*    __restrict__ bias,     // [64]
    const int*      __restrict__ in_idx,   // [27, N]
    const float*    __restrict__ mask,     // [27, N]
    float*          __restrict__ out,      // [N, 64]
    int ntot)
{
    const int tid  = threadIdx.x;
    const int wave = tid >> 5;
    const int lane = tid & 31;
    const int h    = lane >> 4;   // half-wave select
    const int m    = lane & 15;   // row (A/C) or column (B/C) within tile

    const int row_base = blockIdx.x * ROWS_PER_BLOCK + wave * 16;
    const int row      = row_base + m;

    // accumulators: C tile layout is (M = r + 8h, N = nt*16 + m)
    v8f acc[4];
    #pragma unroll
    for (int nt = 0; nt < 4; ++nt) {
        const float bv = bias[nt * 16 + m];
        #pragma unroll
        for (int r = 0; r < 8; ++r) acc[nt][r] = bv;
    }

    // rulebook prefetch pipeline (streamed once -> non-temporal)
    int   irow_n = __builtin_nontemporal_load(&in_idx[(size_t)row]);
    float msk_n  = __builtin_nontemporal_load(&mask  [(size_t)row]);

    const int c0 = h * 8;        // kt=0 first chunk (halfs)
    const int c1 = 32 + h * 8;   // kt=1 first chunk (halfs)

    #pragma unroll 1
    for (int k = 0; k < K_OFF; ++k) {
        const int  irow  = irow_n;
        const bool found = (msk_n != 0.0f);
        if (k + 1 < K_OFF) {
            irow_n = __builtin_nontemporal_load(&in_idx[(size_t)(k + 1) * ntot + row]);
            msk_n  = __builtin_nontemporal_load(&mask  [(size_t)(k + 1) * ntot + row]);
        }

        // wave-uniform skip: whole 16-row tile empty for this offset (~12%)
        if (__builtin_amdgcn_ballot_w32(found) == 0u) continue;

        // mask applied via gather redirection to the zero row
        const int grow = found ? irow : ntot;
        const _Float16* frow = feats16 + (size_t)grow * C_IN;
        const v16h*     bk   = wsB + (size_t)k * 256 + h * 64 + m;

        // A fragments: lane 16h+m holds row m, K chunks [base+8h,+8) U [base+16+8h,+8)
        const v8h a0lo = *(const v8h*)(frow + c0);
        const v8h a0hi = *(const v8h*)(frow + c0 + 16);
        const v8h a1lo = *(const v8h*)(frow + c1);
        const v8h a1hi = *(const v8h*)(frow + c1 + 16);

        // prefetch next offset's gather row (no LOADcnt impact)
        {
            const _Float16* fnext = feats16 + (size_t)irow_n * C_IN;
            __builtin_prefetch(fnext + c0, 0, 3);
            __builtin_prefetch(fnext + c1, 0, 3);
        }

        const v16h a0 = __builtin_shufflevector(a0lo, a0hi,
            0,1,2,3,4,5,6,7,8,9,10,11,12,13,14,15);
        const v16h a1 = __builtin_shufflevector(a1lo, a1hi,
            0,1,2,3,4,5,6,7,8,9,10,11,12,13,14,15);

        #pragma unroll
        for (int nt = 0; nt < 4; ++nt)
            acc[nt] = __builtin_amdgcn_wmma_f32_16x16x32_f16(
                false, a0, false, bk[nt * 16], (short)0, acc[nt], false, false);
        #pragma unroll
        for (int nt = 0; nt < 4; ++nt)
            acc[nt] = __builtin_amdgcn_wmma_f32_16x16x32_f16(
                false, a1, false, bk[128 + nt * 16], (short)0, acc[nt], false, false);
    }

    // store: element (row_base + r + 8h, nt*16 + m); NT keeps feats in L2
    #pragma unroll
    for (int nt = 0; nt < 4; ++nt) {
        #pragma unroll
        for (int r = 0; r < 8; ++r) {
            const size_t orow = (size_t)(row_base + r + 8 * h);
            __builtin_nontemporal_store(acc[nt][r],
                                        &out[orow * C_OUT + nt * 16 + m]);
        }
    }
}

// ---------------------------------------------------------------------------
// Mid fallback (ws >= 221KB): f32 gather + on-the-fly cvt, global B.
// ---------------------------------------------------------------------------
__global__ __launch_bounds__(256) void sparse_conv3d_wmma_gb(
    const float* __restrict__ feats, const v16h* __restrict__ wsB,
    const float* __restrict__ bias, const int* __restrict__ in_idx,
    const float* __restrict__ mask, float* __restrict__ out, int ntot)
{
    const int tid  = threadIdx.x;
    const int wave = tid >> 5;
    const int lane = tid & 31;
    const int h    = lane >> 4;
    const int m    = lane & 15;

    const int row_base = blockIdx.x * ROWS_PER_BLOCK + wave * 16;
    const int row      = row_base + m;

    v8f acc[4];
    #pragma unroll
    for (int nt = 0; nt < 4; ++nt) {
        const float bv = bias[nt * 16 + m];
        #pragma unroll
        for (int r = 0; r < 8; ++r) acc[nt][r] = bv;
    }

    int   irow_n = __builtin_nontemporal_load(&in_idx[(size_t)row]);
    float msk_n  = __builtin_nontemporal_load(&mask  [(size_t)row]);
    const int b0 = h * 8, b1 = 32 + h * 8;

    #pragma unroll 1
    for (int k = 0; k < K_OFF; ++k) {
        const int   irow = irow_n;
        const float msk  = msk_n;
        if (k + 1 < K_OFF) {
            irow_n = __builtin_nontemporal_load(&in_idx[(size_t)(k + 1) * ntot + row]);
            msk_n  = __builtin_nontemporal_load(&mask  [(size_t)(k + 1) * ntot + row]);
        }

        const float* frow = feats + (size_t)irow * C_IN;
        const v16h*  bk   = wsB + (size_t)k * 256 + h * 64 + m;

        const v4f p00 = *(const v4f*)(frow + b0);
        const v4f p01 = *(const v4f*)(frow + b0 + 4);
        const v4f q00 = *(const v4f*)(frow + b0 + 16);
        const v4f q01 = *(const v4f*)(frow + b0 + 20);
        const v4f p10 = *(const v4f*)(frow + b1);
        const v4f p11 = *(const v4f*)(frow + b1 + 4);
        const v4f q10 = *(const v4f*)(frow + b1 + 16);
        const v4f q11 = *(const v4f*)(frow + b1 + 20);

        v16h a0, a1;
        #pragma unroll
        for (int j = 0; j < 4; ++j) {
            a0[j]      = (_Float16)(msk * p00[j]);
            a0[j + 4]  = (_Float16)(msk * p01[j]);
            a0[j + 8]  = (_Float16)(msk * q00[j]);
            a0[j + 12] = (_Float16)(msk * q01[j]);
            a1[j]      = (_Float16)(msk * p10[j]);
            a1[j + 4]  = (_Float16)(msk * p11[j]);
            a1[j + 8]  = (_Float16)(msk * q10[j]);
            a1[j + 12] = (_Float16)(msk * q11[j]);
        }

        #pragma unroll
        for (int nt = 0; nt < 4; ++nt)
            acc[nt] = __builtin_amdgcn_wmma_f32_16x16x32_f16(
                false, a0, false, bk[nt * 16], (short)0, acc[nt], false, false);
        #pragma unroll
        for (int nt = 0; nt < 4; ++nt)
            acc[nt] = __builtin_amdgcn_wmma_f32_16x16x32_f16(
                false, a1, false, bk[128 + nt * 16], (short)0, acc[nt], false, false);
    }

    #pragma unroll
    for (int nt = 0; nt < 4; ++nt) {
        #pragma unroll
        for (int r = 0; r < 8; ++r) {
            const size_t orow = (size_t)(row_base + r + 8 * h);
            __builtin_nontemporal_store(acc[nt][r],
                                        &out[orow * C_OUT + nt * 16 + m]);
        }
    }
}

// ---------------------------------------------------------------------------
// Last fallback (tiny ws): fused per-offset LDS staging.
// ---------------------------------------------------------------------------
__global__ __launch_bounds__(256) void sparse_conv3d_wmma_lds(
    const float* __restrict__ feats, const float* __restrict__ weight,
    const float* __restrict__ bias, const int* __restrict__ in_idx,
    const float* __restrict__ mask, float* __restrict__ out, int ntot)
{
    __shared__ v16h ldsB[256];

    const int tid  = threadIdx.x;
    const int wave = tid >> 5;
    const int lane = tid & 31;
    const int h    = lane >> 4;
    const int m    = lane & 15;

    const int row_base = blockIdx.x * ROWS_PER_BLOCK + wave * 16;
    const int row      = row_base + m;

    v8f acc[4];
    #pragma unroll
    for (int nt = 0; nt < 4; ++nt) {
        const float bv = bias[nt * 16 + m];
        #pragma unroll
        for (int r = 0; r < 8; ++r) acc[nt][r] = bv;
    }

    const int s_kt = (tid >> 7) & 1, s_h = (tid >> 6) & 1, s_n = tid & 63;

    #pragma unroll 1
    for (int k = 0; k < K_OFF; ++k) {
        {
            const float* wk = weight + (size_t)k * (C_IN * C_OUT)
                            + (size_t)(s_kt * 32 + s_h * 8) * C_OUT + s_n;
            v16h wv;
            #pragma unroll
            for (int j = 0; j < 16; ++j) wv[j] = (_Float16)wk[(size_t)(j < 8 ? j : j + 8) * C_OUT];
            ldsB[tid] = wv;
        }
        __syncthreads();

        const int   irow = __builtin_nontemporal_load(&in_idx[(size_t)k * ntot + row]);
        const float msk  = __builtin_nontemporal_load(&mask  [(size_t)k * ntot + row]);
        const float* frow = feats + (size_t)irow * C_IN;

        #pragma unroll
        for (int kt = 0; kt < 2; ++kt) {
            const int base = kt * 32 + h * 8;
            const v4f p0 = *(const v4f*)(frow + base);
            const v4f p1 = *(const v4f*)(frow + base + 4);
            const v4f q0 = *(const v4f*)(frow + base + 16);
            const v4f q1 = *(const v4f*)(frow + base + 20);

            v16h a;
            #pragma unroll
            for (int j = 0; j < 4; ++j) {
                a[j]      = (_Float16)(msk * p0[j]);
                a[j + 4]  = (_Float16)(msk * p1[j]);
                a[j + 8]  = (_Float16)(msk * q0[j]);
                a[j + 12] = (_Float16)(msk * q1[j]);
            }

            #pragma unroll
            for (int nt = 0; nt < 4; ++nt) {
                const v16h b = ldsB[(kt * 2 + h) * 64 + nt * 16 + m];
                acc[nt] = __builtin_amdgcn_wmma_f32_16x16x32_f16(
                    false, a, false, b, (short)0, acc[nt], false, false);
            }
        }
        __syncthreads();
    }

    #pragma unroll
    for (int nt = 0; nt < 4; ++nt) {
        #pragma unroll
        for (int r = 0; r < 8; ++r) {
            const size_t orow = (size_t)(row_base + r + 8 * h);
            __builtin_nontemporal_store(acc[nt][r],
                                        &out[orow * C_OUT + nt * 16 + m]);
        }
    }
}

extern "C" void kernel_launch(void* const* d_in, const int* in_sizes, int n_in,
                              void* d_out, int out_size, void* d_ws, size_t ws_size,
                              hipStream_t stream) {
    const float* feats  = (const float*)d_in[0];
    const float* weight = (const float*)d_in[1];
    const float* bias   = (const float*)d_in[2];
    const int*   in_idx = (const int*)  d_in[3];
    // d_in[4] = out_idx: identity wherever mask==1, unused by construction
    const float* mask   = (const float*)d_in[5];
    float* out = (float*)d_out;

    const int ntot   = in_sizes[0] / C_IN;          // 262144
    const int blocks = ntot / ROWS_PER_BLOCK;       // 2048

    const size_t f16_bytes  = (size_t)(ntot + 1) * C_IN * sizeof(_Float16);
    const size_t ws_full    = WSB_BYTES + f16_bytes;   // ~33.8 MB

    if (ws_size >= ws_full) {
        v16h*     wsB = (v16h*)d_ws;
        _Float16* f16 = (_Float16*)((char*)d_ws + WSB_BYTES);
        convert_weights_bswizzle<<<K_OFF, 256, 0, stream>>>(weight, wsB);
        const int cvt_blocks = (int)(((size_t)ntot * C_IN / 8 + 255) / 256);
        convert_feats_f16<<<cvt_blocks, 256, 0, stream>>>(feats, f16, ntot);
        sparse_conv3d_wmma_f16tab<<<blocks, 256, 0, stream>>>(
            f16, wsB, bias, in_idx, mask, out, ntot);
    } else if (ws_size >= WSB_BYTES) {
        v16h* wsB = (v16h*)d_ws;
        convert_weights_bswizzle<<<K_OFF, 256, 0, stream>>>(weight, wsB);
        sparse_conv3d_wmma_gb<<<blocks, 256, 0, stream>>>(
            feats, wsB, bias, in_idx, mask, out, ntot);
    } else {
        sparse_conv3d_wmma_lds<<<blocks, 256, 0, stream>>>(
            feats, weight, bias, in_idx, mask, out, ntot);
    }
}
